// DeepVtx_58497454571627
// MI455X (gfx1250) — matrix operand order repaired and checked by reading, not compiled, not run on hardware
//
#include <hip/hip_runtime.h>
#include <hip/hip_bf16.h>
#include <cstdint>
#include <cstddef>

// ============================================================================
// DeepVtx sparse U-Net forward for MI455X (gfx1250, wave32, WMMA).
//
// All convs use V_WMMA_F32_16X16X32_F16: activations are BN-ReLU'd and cast
// to f16 (padded to Cin multiple of 32), weights are pre-packed into the
// contiguous-k B-fragment lane layout (same k order as A), accumulation is f32. Each wave keeps
// NBW (up to 8) accumulator tiles so every gathered A fragment feeds up to
// 8 WMMAs. Rulebook entries are preloaded 4 taps at a time and the gather
// rows prefetched. NOTE: an explicit B-fragment batching + inline-asm fence
// was tried and reverted — it forced accumulator spills to scratch, which
// costs more than the per-WMMA L0-hit wait it removes (hidden by occupancy).
//
// Input flattening (setup_inputs() insertion order, tuples in order):
//   0: feats [N0,1]
//   1: W_in  [27,1,16]
//   level l in 0..3 at base = 2 + 18*l:
//     +0..2  pre0 (g,b,W[27,P,P])   +3..5  pre1 (g,b,W)
//     +6..7  down_bn (g,b)          +8     down_W [8,P,2P]
//     +9..10 up_bn (g,b)            +11    up_W   [8,2P,P]
//     +12..14 post0 (g,b,W[27,2P,P]) +15..17 post1 (g,b,W[27,P,P])
//   level 4 at 74: pre0 (g,b,W[27,256,256]) 74..76, pre1 77..79
//   80 final_g  81 final_b  82 lin_W[16]  83 lin_b[1]
//   84+l : sm_l [N_l,27]   (l=0..4)
//   89+3l: down_l [N_{l+1},8]  90+3l: parent_l [N_l]  91+3l: kidx_l [N_l]
// ============================================================================

typedef __attribute__((ext_vector_type(16))) _Float16 v16h;
typedef __attribute__((ext_vector_type(8)))  _Float16 v8h;
typedef __attribute__((ext_vector_type(8)))  float    v8f;

#define BN_EPS 1e-4f

// ---------------------------------------------------------------------------
// Deterministic, coalesced BN statistics.
// Stage 1: linear grid-stride over x (row-major [M,C]). chunk % 256 == 0 and
// C | 256 (C in {16,32,64,128,256}), so each thread's elements all belong to
// channel t % C -> private accumulation, then deterministic LDS tree reduce
// over the 256/C threads sharing a channel. Partials per block are written
// in fixed order; stage 2 reduces blocks in fixed order. No float atomics.
__global__ void dv_bn_partials(const float* __restrict__ x, float* __restrict__ ps,
                               float* __restrict__ pss, long long E, int C,
                               long long chunk) {
    const long long base = (long long)blockIdx.x * chunk;
    const long long end  = (base + chunk < E) ? base + chunk : E;
    const int t = threadIdx.x;
    float s = 0.f, ss = 0.f;
    for (long long i = base + t; i < end; i += 256) {
        float v = x[i];
        s += v; ss += v * v;
    }
    __shared__ float ls[256], lss[256];
    ls[t] = s; lss[t] = ss;
    __syncthreads();
    for (int o = 128; o >= C; o >>= 1) {          // o is always a multiple of C
        if (t < o) { ls[t] += ls[t + o]; lss[t] += lss[t + o]; }
        __syncthreads();
    }
    if (t < C) {
        ps [(long long)blockIdx.x * C + t] = ls[t];
        pss[(long long)blockIdx.x * C + t] = lss[t];
    }
}

__global__ void dv_bn_reduce(const float* __restrict__ ps, const float* __restrict__ pss,
                             const float* __restrict__ g, const float* __restrict__ b,
                             float* __restrict__ scale, float* __restrict__ shift,
                             int B, int C, int M) {
    int c = blockIdx.x * blockDim.x + threadIdx.x;
    if (c >= C) return;
    float s = 0.f, ss = 0.f;
    for (int i = 0; i < B; ++i) {
        s  += ps [(long long)i * C + c];
        ss += pss[(long long)i * C + c];
    }
    float inv = 1.0f / (float)M;
    float mu  = s * inv;
    float var = ss * inv - mu * mu;
    float sc  = g[c] * rsqrtf(var + BN_EPS);
    scale[c] = sc;
    shift[c] = b[c] - mu * sc;
}

// ---------------------------------------------------------------------------
// Fused BN + ReLU + cast to f16 with zero-padding of channels to Cpad.
__global__ void dv_cast(const float* __restrict__ x, _Float16* __restrict__ xh,
                        const float* __restrict__ scale, const float* __restrict__ shift,
                        int M, int C, int Cpad, int doRelu) {
    long long i = (long long)blockIdx.x * blockDim.x + threadIdx.x;
    long long total = (long long)M * Cpad;
    if (i >= total) return;
    int c = (int)(i % Cpad);
    long long m = i / Cpad;
    float v = 0.f;
    if (c < C) {
        v = x[m * C + c];
        if (scale) v = v * scale[c] + shift[c];
        if (doRelu) v = fmaxf(v, 0.f);
    }
    xh[i] = (_Float16)v;
}

// ---------------------------------------------------------------------------
// Pack W[k][cin][cout] (f32) into contiguous-k B-fragment lane order (f16).
// Fragment (k, cb, nb): 32 lanes x 16 halfs; lane column = lane&15,
// K element e maps to cin = cb*32 + (lane>>4)*16 + e  (lanes 0-15: K 0-15,
// lanes 16-31: K 16-31 same order as the A gather).
__global__ void dv_pack(const float* __restrict__ W, _Float16* __restrict__ wp,
                        int K, int Cin, int Cpad, int Cout) {
    int t = blockIdx.x * blockDim.x + threadIdx.x;
    int lane = t & 31;
    int frag = t >> 5;
    int Cin32 = Cpad >> 5;
    int NB = Cout >> 4;
    int total = K * Cin32 * NB;
    if (frag >= total) return;
    int nb = frag % NB;
    int tmp = frag / NB;
    int cb = tmp % Cin32;
    int k  = tmp / Cin32;
    int col = lane & 15, hi = lane >> 4;
    int cout = nb * 16 + col;
    _Float16* dst = wp + (long long)frag * 512 + lane * 16;
#pragma unroll
    for (int e = 0; e < 16; ++e) {
        int cin = cb * 32 + hi * 16 + e;
        float v = (cin < Cin) ? W[((long long)k * Cin + cin) * Cout + cout] : 0.f;
        dst[e] = (_Float16)v;
    }
}

// ---------------------------------------------------------------------------
// Generic rulebook gather conv with NBW accumulator tiles per wave:
//   out[m, off + :NBW*16] = sum_k xh[nbr[m,k]] * Wp[k]
// One wave per (16-site tile, NBW*16 cout slab). Rulebook entries preloaded
// 4 taps/group (constant indices -> stay in VGPRs); empty taps skipped
// wave-uniformly via ballot; gather rows prefetched per group.
template <int NBW>
__global__ void dv_conv_t(const _Float16* __restrict__ xh,   // [*, Cpad] f16
                          const _Float16* __restrict__ wp,   // packed fragments
                          const int* __restrict__ nbr,       // [Mout, K]
                          float* __restrict__ out,
                          int Mout, int K, int Cpad, int Cout,
                          int outStride, int outOff) {
    const int lane = threadIdx.x & 31;
    const int wave = threadIdx.x >> 5;
    const int tile = blockIdx.x * (blockDim.x >> 5) + wave;
    const int row0 = tile * 16;
    if (row0 >= Mout) return;                 // wave-uniform exit
    const int nb0 = blockIdx.y * NBW;
    const int col = lane & 15;
    const int hi  = lane >> 4;
    const int site = row0 + col;
    const bool siteOk = (site < Mout);
    const int Cin32 = Cpad >> 5;
    const int NB = Cout >> 4;

    const v8f vzero = {};
    v8f acc[NBW];
#pragma unroll
    for (int j = 0; j < NBW; ++j) acc[j] = vzero;

    for (int k0 = 0; k0 < K; k0 += 4) {
        // Preload up to 4 rulebook entries (one clause, one wait) and
        // prefetch their gather rows before any body executes.
        int nk[4];
#pragma unroll
        for (int u = 0; u < 4; ++u) {
            int k = k0 + u;
            nk[u] = (k < K && siteOk) ? nbr[(long long)site * K + k] : -1;
        }
#pragma unroll
        for (int u = 0; u < 4; ++u) {
            if (nk[u] >= 0)
                __builtin_prefetch(xh + (long long)nk[u] * Cpad + hi * 8, 0, 1);
        }
#pragma unroll
        for (int u = 0; u < 4; ++u) {
            const int n = nk[u];
            if (__builtin_amdgcn_ballot_w32(n >= 0) == 0u) continue;  // tap empty
            const _Float16* xrow = xh + (long long)n * Cpad;
            for (int cb = 0; cb < Cin32; ++cb) {
                // A fragment: row = lane&15; K elems e0..15 -> hi*16 + (0..15)
                // (contiguous, matching packed B layout).
                v16h A;
#pragma unroll
                for (int i = 0; i < 16; ++i) A[i] = (_Float16)0.f;
                if (n >= 0) {
                    const _Float16* p = xrow + cb * 32 + hi * 16;
                    v8h a0 = *(const v8h*)(p);
                    v8h a1 = *(const v8h*)(p + 8);
#pragma unroll
                    for (int i = 0; i < 8; ++i) { A[i] = a0[i]; A[8 + i] = a1[i]; }
                }
                // NBW B fragments reuse the same A: independent-D WMMA chain.
                const _Float16* wf0 =
                    wp + (((long long)((k0 + u) * Cin32 + cb) * NB + nb0) * 512)
                       + lane * 16;
#pragma unroll
                for (int j = 0; j < NBW; ++j) {
                    v8h b0 = *(const v8h*)(wf0 + (long long)j * 512);
                    v8h b1 = *(const v8h*)(wf0 + (long long)j * 512 + 8);
                    v16h Bv;
#pragma unroll
                    for (int i = 0; i < 8; ++i) { Bv[i] = b0[i]; Bv[8 + i] = b1[i]; }
                    acc[j] = __builtin_amdgcn_wmma_f32_16x16x32_f16(
                        false, A, false, Bv, (short)0, acc[j], false, false);
                }
            }
        }
    }
    // D layout: element v -> row = hi*8 + v, column = lane&15.
#pragma unroll
    for (int j = 0; j < NBW; ++j) {
#pragma unroll
        for (int v = 0; v < 8; ++v) {
            int r = row0 + hi * 8 + v;
            if (r < Mout)
                out[(long long)r * outStride + outOff + (nb0 + j) * 16 + col] = acc[j][v];
        }
    }
}

// ---------------------------------------------------------------------------
__global__ void dv_copy(const float* __restrict__ src, float* __restrict__ dst,
                        int M, int C, int dstStride, int dstOff) {
    long long i = (long long)blockIdx.x * blockDim.x + threadIdx.x;
    if (i >= (long long)M * C) return;
    int c = (int)(i % C);
    long long m = i / C;
    dst[m * dstStride + dstOff + c] = src[i];
}

// Deconv as 8-tap gather: dnbr[n][k] = (kidx[n]==k) ? parent[n] : -1
__global__ void dv_dnbr(const int* __restrict__ parent, const int* __restrict__ kidx,
                        int* __restrict__ dnbr, int M) {
    int n = blockIdx.x * blockDim.x + threadIdx.x;
    if (n >= M) return;
    int p = parent[n], ki = kidx[n];
#pragma unroll
    for (int k = 0; k < 8; ++k) dnbr[n * 8 + k] = (k == ki) ? p : -1;
}

// Final BN-ReLU + linear(16->1) + sigmoid.
__global__ void dv_head(const float* __restrict__ x, const float* __restrict__ scale,
                        const float* __restrict__ shift, const float* __restrict__ linW,
                        const float* __restrict__ linb, float* __restrict__ out, int M) {
    int n = blockIdx.x * blockDim.x + threadIdx.x;
    if (n >= M) return;
    float acc = linb[0];
#pragma unroll
    for (int c = 0; c < 16; ++c) {
        float v = x[(long long)n * 16 + c] * scale[c] + shift[c];
        v = fmaxf(v, 0.f);
        acc += v * linW[c];
    }
    out[n] = 1.f / (1.f + expf(-acc));
}

// ===========================================================================
// Host orchestration
// ===========================================================================

static inline int cdiv_i(long long a, long long b) { return (int)((a + b - 1) / b); }
static inline int pad32(int c) { return (c + 31) & ~31; }

static const int PL[5] = {16, 32, 64, 128, 256};

struct NetCtx {
    hipStream_t stream;
    void* const* d_in;
    int N[5];
    float* xlev[5];
    float* cbuf;
    int*   dnbr;
    float *ps, *pss, *scale, *shift;
    _Float16 *xh, *wp;
    const float* F(int i) const { return (const float*)d_in[i]; }
    const int*   I(int i) const { return (const int*)d_in[i]; }
};

static void run_bn(NetCtx& C, const float* x, int M, int Cn,
                   const float* g, const float* b) {
    long long E = (long long)M * Cn;
    long long chunk = (E + 511) / 512;          // <= 512 blocks
    chunk = (chunk + 255) & ~255LL;
    if (chunk < 256) chunk = 256;
    int B = (int)((E + chunk - 1) / chunk);
    dv_bn_partials<<<B, 256, 0, C.stream>>>(x, C.ps, C.pss, E, Cn, chunk);
    dv_bn_reduce<<<cdiv_i(Cn, 256), 256, 0, C.stream>>>(C.ps, C.pss, g, b,
                                                        C.scale, C.shift, B, Cn, M);
}

// BN(optional) -> cast f16 -> pack weights -> WMMA gather conv.
static void run_conv(NetCtx& C, const float* x, int Min, int Cin,
                     const int* nbr, int K, int Mout,
                     const float* W, int Cout,
                     float* out, int outStride, int outOff,
                     const float* g, const float* b) {
    hipStream_t s = C.stream;
    const float* sc = nullptr; const float* sh = nullptr; int relu = 0;
    if (g) {
        run_bn(C, x, Min, Cin, g, b);
        sc = C.scale; sh = C.shift; relu = 1;
    }
    int Cp = pad32(Cin);
    long long tot = (long long)Min * Cp;
    dv_cast<<<cdiv_i(tot, 256), 256, 0, s>>>(x, C.xh, sc, sh, Min, Cin, Cp, relu);
    int Cin32 = Cp / 32, NB = Cout / 16;
    long long frags = (long long)K * Cin32 * NB;
    dv_pack<<<cdiv_i(frags * 32, 256), 256, 0, s>>>(W, C.wp, K, Cin, Cp, Cout);
    int nbw = (NB % 8 == 0) ? 8 : (NB % 4 == 0) ? 4 : (NB % 2 == 0) ? 2 : 1;
    dim3 grid((unsigned)cdiv_i(Mout, 128), (unsigned)(NB / nbw));  // 8 waves * 16 rows
    switch (nbw) {
    case 8: dv_conv_t<8><<<grid, 256, 0, s>>>(C.xh, C.wp, nbr, out, Mout, K, Cp, Cout,
                                              outStride, outOff); break;
    case 4: dv_conv_t<4><<<grid, 256, 0, s>>>(C.xh, C.wp, nbr, out, Mout, K, Cp, Cout,
                                              outStride, outOff); break;
    case 2: dv_conv_t<2><<<grid, 256, 0, s>>>(C.xh, C.wp, nbr, out, Mout, K, Cp, Cout,
                                              outStride, outOff); break;
    default: dv_conv_t<1><<<grid, 256, 0, s>>>(C.xh, C.wp, nbr, out, Mout, K, Cp, Cout,
                                               outStride, outOff); break;
    }
}

static void unet(NetCtx& C, int l) {
    const int M = C.N[l], P = PL[l];
    const int base = (l < 4) ? (2 + 18 * l) : 74;
    const int* sm = C.I(84 + l);
    float* x = C.xlev[l];
    // pre convs (BN-ReLU -> 27-tap submanifold conv), in-place via f16 staging
    run_conv(C, x, M, P, sm, 27, M, C.F(base + 2), P, x, P, 0, C.F(base + 0), C.F(base + 1));
    run_conv(C, x, M, P, sm, 27, M, C.F(base + 5), P, x, P, 0, C.F(base + 3), C.F(base + 4));
    if (l == 4) return;
    const int Q = PL[l + 1], Mc = C.N[l + 1];
    // downsample: BN over fine x, gather 8 fine taps per coarse site
    run_conv(C, x, M, P, C.I(89 + 3 * l), 8, Mc, C.F(base + 8), Q,
             C.xlev[l + 1], Q, 0, C.F(base + 6), C.F(base + 7));
    unet(C, l + 1);
    // deconv into concat buffer second half; copy x into first half
    dv_dnbr<<<cdiv_i(M, 256), 256, 0, C.stream>>>(C.I(90 + 3 * l), C.I(91 + 3 * l), C.dnbr, M);
    dv_copy<<<cdiv_i((long long)M * P, 256), 256, 0, C.stream>>>(x, C.cbuf, M, P, 2 * P, 0);
    run_conv(C, C.xlev[l + 1], Mc, Q, C.dnbr, 8, M, C.F(base + 11), P,
             C.cbuf, 2 * P, P, C.F(base + 9), C.F(base + 10));
    // post convs
    run_conv(C, C.cbuf, M, 2 * P, sm, 27, M, C.F(base + 14), P, x, P, 0,
             C.F(base + 12), C.F(base + 13));
    run_conv(C, x, M, P, sm, 27, M, C.F(base + 17), P, x, P, 0,
             C.F(base + 15), C.F(base + 16));
}

extern "C" void kernel_launch(void* const* d_in, const int* in_sizes, int n_in,
                              void* d_out, int out_size, void* d_ws, size_t ws_size,
                              hipStream_t stream) {
    (void)n_in; (void)out_size; (void)ws_size;
    NetCtx C;
    C.stream = stream;
    C.d_in = d_in;
    C.N[0] = in_sizes[0];
    for (int l = 1; l < 5; ++l) C.N[l] = in_sizes[84 + l] / 27;

    // Workspace carve (256B aligned cursor).
    uint8_t* wsb = (uint8_t*)d_ws;
    size_t cur = 0;
    auto take = [&](size_t bytes) -> void* {
        cur = (cur + 255) & ~(size_t)255;
        void* p = wsb + cur;
        cur += bytes;
        return p;
    };
    C.ps    = (float*)take((size_t)512 * 256 * sizeof(float));
    C.pss   = (float*)take((size_t)512 * 256 * sizeof(float));
    C.scale = (float*)take(512 * sizeof(float));
    C.shift = (float*)take(512 * sizeof(float));
    for (int l = 0; l < 5; ++l)
        C.xlev[l] = (float*)take((size_t)C.N[l] * PL[l] * sizeof(float));
    size_t cb = 0;
    for (int l = 0; l < 4; ++l) {
        size_t s = (size_t)C.N[l] * 2 * PL[l] * sizeof(float);
        if (s > cb) cb = s;
    }
    C.cbuf = (float*)take(cb);           // concat buffers: lifetimes disjoint -> shared
    size_t mx = (size_t)C.N[0] * 32;     // f16 staging: max M * pad32(Cin) over all convs
    {
        size_t cand[4] = {(size_t)C.N[1] * 64, (size_t)C.N[2] * 128,
                          (size_t)C.N[3] * 256, (size_t)C.N[4] * 256};
        for (int i = 0; i < 4; ++i) if (cand[i] > mx) mx = cand[i];
    }
    C.xh = (_Float16*)take(mx * sizeof(_Float16));
    C.wp = (_Float16*)take((size_t)27 * 8 * 16 * 512 * sizeof(_Float16)); // L4 pre = max
    C.dnbr = (int*)take((size_t)C.N[0] * 8 * sizeof(int));

    // Input submanifold conv 1->16 (no BN, no ReLU; cin padded to 32).
    run_conv(C, C.F(0), C.N[0], 1, C.I(84), 27, C.N[0], C.F(1), 16,
             C.xlev[0], 16, 0, nullptr, nullptr);
    unet(C, 0);

    // Head: final BN-ReLU + linear + sigmoid.
    run_bn(C, C.xlev[0], C.N[0], 16, C.F(80), C.F(81));
    dv_head<<<cdiv_i(C.N[0], 256), 256, 0, stream>>>(C.xlev[0], C.scale, C.shift,
                                                     C.F(82), C.F(83),
                                                     (float*)d_out, C.N[0]);
}
